// PointHR_53687091200711
// MI455X (gfx1250) — matrix operand order repaired and checked by reading, not compiled, not run on hardware
//
#include <hip/hip_runtime.h>

// ---------------------------------------------------------------------------
// PointTransformer-style grouped vector attention for MI455X (gfx1250).
// wave32, WMMA f32_16x16x32_f16 for all GEMMs. See analysis for pass design.
// ---------------------------------------------------------------------------

typedef __attribute__((ext_vector_type(16))) _Float16     v16h;
typedef __attribute__((ext_vector_type(8)))  float        v8f;
typedef __attribute__((ext_vector_type(4)))  unsigned int u32x4;

#define NPTS 65536
#define CCH  256
#define GRP  16
#define SNB  16
#define EPSV 1e-5f

union AB16 { v16h v; u32x4 q[2]; };

// A fragment: 16x32 f16, rows m0.., K in [k0,k0+32), row stride ld (halves).
// Lane l: m = m0 + (l&15), kb = l>>4. halves 0..7 -> K = k0+kb*8+0..7,
// halves 8..15 -> K = k0+kb*8+16..23  (ISA 16-bit A 16x32 table).
__device__ inline v16h load_frag_A(const _Float16* p, int m0, int k0, int ld, int lane) {
    int m  = m0 + (lane & 15);
    int kb = lane >> 4;
    const _Float16* base = p + (size_t)m * ld + k0 + kb * 8;
    AB16 f;
    f.q[0] = *(const u32x4*)(base);
    f.q[1] = *(const u32x4*)(base + 16);
    return f.v;
}

// B fragment: 32x16 f16 from TRANSPOSED weights Wt[N][K] row-major.
// Lane l: n = n0 + (l&15), kb = l>>4. halves 0..7 -> K = k0+kb*8+0..7,
// halves 8..15 -> K = k0+kb*8+16..23 (same table order as A).
__device__ inline v16h load_frag_B(const _Float16* wt, int n0, int k0, int ld, int lane) {
    int n  = n0 + (lane & 15);
    int kb = lane >> 4;
    const _Float16* base = wt + (size_t)n * ld + k0 + kb * 8;
    AB16 f;
    f.q[0] = *(const u32x4*)(base);
    f.q[1] = *(const u32x4*)(base + 16);
    return f.v;
}

__device__ inline v8f wmma_f16(v16h a, v16h b, v8f c) {
    return __builtin_amdgcn_wmma_f32_16x16x32_f16(false, a, false, b, (short)0, c, false, false);
}

// ---------------------------------------------------------------------------
// prep: feat -> f16, weights -> transposed f16 (contiguous-K B fragments)
// ---------------------------------------------------------------------------
__global__ __launch_bounds__(256) void prep_kernel(
    const float* __restrict__ feat,
    const float* __restrict__ Wq, const float* __restrict__ Wk,
    const float* __restrict__ Wv, const float* __restrict__ Wp2,
    const float* __restrict__ Ww1,
    _Float16* __restrict__ feat16,
    _Float16* __restrict__ WqT, _Float16* __restrict__ WkT,
    _Float16* __restrict__ WvT, _Float16* __restrict__ Wp2T,
    _Float16* __restrict__ Ww1T)
{
    size_t i = (size_t)blockIdx.x * 256 + threadIdx.x;
    feat16[i] = (_Float16)feat[i];                       // grid == N*C exactly
    if (i < (size_t)CCH * CCH) {
        int kk = (int)(i / CCH), n = (int)(i % CCH);
        WqT [n * CCH + kk] = (_Float16)Wq [kk * CCH + n];
        WkT [n * CCH + kk] = (_Float16)Wk [kk * CCH + n];
        WvT [n * CCH + kk] = (_Float16)Wv [kk * CCH + n];
        Wp2T[n * CCH + kk] = (_Float16)Wp2[kk * CCH + n];
    }
    if (i < (size_t)CCH * GRP) {
        int kk = (int)(i / GRP), g = (int)(i % GRP);
        Ww1T[g * CCH + kk] = (_Float16)Ww1[kk * GRP + g];
    }
}

// ---------------------------------------------------------------------------
// q/k/v = feat @ W* + b*   (WMMA, 128-row tile per block, 8 waves)
// ---------------------------------------------------------------------------
__global__ __launch_bounds__(256) void gemm_qkv_kernel(
    const _Float16* __restrict__ feat16,
    const _Float16* __restrict__ WqT, const _Float16* __restrict__ WkT,
    const _Float16* __restrict__ WvT,
    const float* __restrict__ bq, const float* __restrict__ bk,
    const float* __restrict__ bv,
    float* __restrict__ q_lin, float* __restrict__ k_lin,
    float* __restrict__ v_out)
{
    int wave = threadIdx.x >> 5;
    int lane = threadIdx.x & 31;
    int row0 = blockIdx.x * 128 + wave * 16;
    int ncol = lane & 15, hi = lane >> 4;

    v16h a[8];
    for (int kk = 0; kk < 8; ++kk)
        a[kk] = load_frag_A(feat16, row0, kk * 32, CCH, lane);

    const _Float16* WT[3]  = { WqT, WkT, WvT };
    const float*    bias[3] = { bq, bk, bv };
    float*          outp[3] = { q_lin, k_lin, v_out };

    for (int o = 0; o < 3; ++o) {
        for (int nt = 0; nt < 16; ++nt) {
            int n0 = nt * 16;
            float bval = bias[o][n0 + ncol];
            v8f acc;
            for (int r = 0; r < 8; ++r) acc[r] = bval;
            for (int kk = 0; kk < 8; ++kk) {
                v16h b = load_frag_B(WT[o], n0, kk * 32, CCH, lane);
                acc = wmma_f16(a[kk], b, acc);
            }
            float* op = outp[o];
            int n = n0 + ncol;
            for (int r = 0; r < 8; ++r)
                op[(size_t)(row0 + r + 8 * hi) * CCH + n] = acc[r];
        }
    }
}

// ---------------------------------------------------------------------------
// per-channel sum/sumsq of q_lin,k_lin (for BatchNorm training stats)
// ---------------------------------------------------------------------------
__global__ __launch_bounds__(256) void channel_stats_kernel(
    const float* __restrict__ q_lin, const float* __restrict__ k_lin,
    float* __restrict__ stats)
{
    int c = threadIdx.x;
    size_t row0 = (size_t)blockIdx.x * 256;
    float s1q = 0.f, s2q = 0.f, s1k = 0.f, s2k = 0.f;
    for (int i = 0; i < 256; ++i) {
        float vq = q_lin[(row0 + i) * CCH + c];
        float vk = k_lin[(row0 + i) * CCH + c];
        s1q += vq; s2q += vq * vq;
        s1k += vk; s2k += vk * vk;
    }
    atomicAdd(&stats[c],        s1q);
    atomicAdd(&stats[256 + c],  s2q);
    atomicAdd(&stats[512 + c],  s1k);
    atomicAdd(&stats[768 + c],  s2k);
}

// ---------------------------------------------------------------------------
// pos = coord[idx] - coord : 3-vector mean + 3x3 second moment (9 sums)
// ---------------------------------------------------------------------------
__global__ __launch_bounds__(256) void pos_stats_kernel(
    const float* __restrict__ coord, const int* __restrict__ ref,
    float* __restrict__ pstats)
{
    __shared__ float acc[9];
    if (threadIdx.x < 9) acc[threadIdx.x] = 0.f;
    __syncthreads();
    size_t id = (size_t)blockIdx.x * 256 + threadIdx.x;
    int n = (int)(id >> 4);
    int s = (int)(id & 15);
    int r = ref[(size_t)n * SNB + s];
    int idx = r < 0 ? 0 : r;
    float px = coord[(size_t)idx * 3 + 0] - coord[(size_t)n * 3 + 0];
    float py = coord[(size_t)idx * 3 + 1] - coord[(size_t)n * 3 + 1];
    float pz = coord[(size_t)idx * 3 + 2] - coord[(size_t)n * 3 + 2];
    atomicAdd(&acc[0], px);      atomicAdd(&acc[1], py);      atomicAdd(&acc[2], pz);
    atomicAdd(&acc[3], px * px); atomicAdd(&acc[4], px * py); atomicAdd(&acc[5], px * pz);
    atomicAdd(&acc[6], py * py); atomicAdd(&acc[7], py * pz); atomicAdd(&acc[8], pz * pz);
    __syncthreads();
    if (threadIdx.x < 9) atomicAdd(&pstats[threadIdx.x], acc[threadIdx.x]);
}

// ---------------------------------------------------------------------------
// fold BN stats into per-channel affine (aQ,bQ,aK,bK, aP,cP)
// positional BN variance computed analytically from 3x3 covariance of pos.
// ---------------------------------------------------------------------------
__global__ void finalize_bn_kernel(
    const float* __restrict__ stats, const float* __restrict__ pstats,
    const float* __restrict__ gq, const float* __restrict__ Bq,
    const float* __restrict__ gk, const float* __restrict__ Bk,
    const float* __restrict__ Wp1, const float* __restrict__ gp,
    const float* __restrict__ Bp,
    float* __restrict__ bnp)
{
    int c = threadIdx.x;
    if (c >= CCH) return;
    float invN = 1.f / (float)NPTS;
    float mq = stats[c] * invN;
    float vq = stats[256 + c] * invN - mq * mq;
    float aQ = gq[c] * rsqrtf(vq + EPSV);
    bnp[c]        = aQ;
    bnp[256 + c]  = Bq[c] - mq * aQ;
    float mk = stats[512 + c] * invN;
    float vk = stats[768 + c] * invN - mk * mk;
    float aK = gk[c] * rsqrtf(vk + EPSV);
    bnp[512 + c]  = aK;
    bnp[768 + c]  = Bk[c] - mk * aK;
    // positional BN (bias bp1 cancels inside (y-mean))
    float cnt = (float)NPTS * (float)SNB;
    float m0 = pstats[0] / cnt, m1 = pstats[1] / cnt, m2 = pstats[2] / cnt;
    float Cxx = pstats[3] / cnt - m0 * m0;
    float Cxy = pstats[4] / cnt - m0 * m1;
    float Cxz = pstats[5] / cnt - m0 * m2;
    float Cyy = pstats[6] / cnt - m1 * m1;
    float Cyz = pstats[7] / cnt - m1 * m2;
    float Czz = pstats[8] / cnt - m2 * m2;
    float w0 = Wp1[c], w1 = Wp1[CCH + c], w2 = Wp1[2 * CCH + c];
    float mu  = m0 * w0 + m1 * w1 + m2 * w2;
    float var = w0 * w0 * Cxx + w1 * w1 * Cyy + w2 * w2 * Czz
              + 2.f * (w0 * w1 * Cxy + w0 * w2 * Cxz + w1 * w2 * Cyz);
    float aP = gp[c] * rsqrtf(var + EPSV);
    bnp[1024 + c] = aP;
    bnp[1280 + c] = Bp[c] - mu * aP;
}

__global__ __launch_bounds__(256) void apply_bn_kernel(
    float* __restrict__ q, float* __restrict__ k, const float* __restrict__ bnp)
{
    size_t i = (size_t)blockIdx.x * 256 + threadIdx.x;
    int c = (int)(i & 255);
    float x = q[i] * bnp[c] + bnp[256 + c];
    q[i] = x > 0.f ? x : 0.f;
    float y = k[i] * bnp[512 + c] + bnp[768 + c];
    k[i] = y > 0.f ? y : 0.f;
}

// ---------------------------------------------------------------------------
// Pass 2: per point -> pe1 -> peb (WMMA) -> r = k[idx]-q+peb -> wpre = r@Ww1
// also accumulates global sum/sumsq of wpre per group channel.
// ---------------------------------------------------------------------------
__global__ __launch_bounds__(256) void wpre_kernel(
    const float* __restrict__ coord, const int* __restrict__ ref,
    const float* __restrict__ q, const float* __restrict__ k,
    const float* __restrict__ Wp1, const float* __restrict__ bp2,
    const _Float16* __restrict__ Wp2T, const _Float16* __restrict__ Ww1T,
    const float* __restrict__ bw1, const float* __restrict__ bnp,
    float* __restrict__ wpre, float* __restrict__ wstats)
{
    const int LDL = 264;
    __shared__ __align__(16) _Float16 s_mat[16 * LDL];  // pe1, then r
    __shared__ float s_part[8 * 256];
    __shared__ float s_pos[16][3];
    __shared__ int   s_idx[16];
    __shared__ float s_sum[16], s_ssq[16];

    int t = threadIdx.x;
    int point = blockIdx.x;
    if (t < 16) {
        int r = ref[(size_t)point * SNB + t];
        int idx = r < 0 ? 0 : r;
        s_idx[t] = idx;
        s_sum[t] = 0.f; s_ssq[t] = 0.f;
        float cx = coord[(size_t)point * 3 + 0];
        float cy = coord[(size_t)point * 3 + 1];
        float cz = coord[(size_t)point * 3 + 2];
        s_pos[t][0] = coord[(size_t)idx * 3 + 0] - cx;
        s_pos[t][1] = coord[(size_t)idx * 3 + 1] - cy;
        s_pos[t][2] = coord[(size_t)idx * 3 + 2] - cz;
    }
    __syncthreads();

    {   // pe1[s][c] = relu((pos . Wp1[:,c]) * aP + cP), one channel per thread
        int c = t;
        float w0 = Wp1[c], w1 = Wp1[CCH + c], w2 = Wp1[2 * CCH + c];
        float a = bnp[1024 + c], b = bnp[1280 + c];
        for (int s = 0; s < SNB; ++s) {
            float z = s_pos[s][0] * w0 + s_pos[s][1] * w1 + s_pos[s][2] * w2;
            float val = z * a + b;
            s_mat[s * LDL + c] = (_Float16)(val > 0.f ? val : 0.f);
        }
    }
    __syncthreads();

    int wave = t >> 5, lane = t & 31;
    int ncol = lane & 15, hi = lane >> 4;

    v16h a[8];
    for (int kk = 0; kk < 8; ++kk)
        a[kk] = load_frag_A(s_mat, 0, kk * 32, LDL, lane);

    int n0a = wave * 32, n0b = n0a + 16;
    v8f acc0, acc1;
    { float bv = bp2[n0a + ncol]; for (int r = 0; r < 8; ++r) acc0[r] = bv; }
    { float bv = bp2[n0b + ncol]; for (int r = 0; r < 8; ++r) acc1[r] = bv; }
    for (int kk = 0; kk < 8; ++kk) {
        acc0 = wmma_f16(a[kk], load_frag_B(Wp2T, n0a, kk * 32, CCH, lane), acc0);
        acc1 = wmma_f16(a[kk], load_frag_B(Wp2T, n0b, kk * 32, CCH, lane), acc1);
    }
    __syncthreads();   // everyone done reading pe1; reuse s_mat for r

    {   // r = peb + k[idx] - q[point]  (f16 into LDS)
        int n = n0a + ncol;
        float qn = q[(size_t)point * CCH + n];
        for (int r = 0; r < 8; ++r) {
            int m = r + 8 * hi;
            float rv = acc0[r] + k[(size_t)s_idx[m] * CCH + n] - qn;
            s_mat[m * LDL + n] = (_Float16)rv;
        }
        n = n0b + ncol;
        qn = q[(size_t)point * CCH + n];
        for (int r = 0; r < 8; ++r) {
            int m = r + 8 * hi;
            float rv = acc1[r] + k[(size_t)s_idx[m] * CCH + n] - qn;
            s_mat[m * LDL + n] = (_Float16)rv;
        }
    }
    __syncthreads();

    {   // wpre = r @ Ww1 : K=256 split across 8 waves, one WMMA each
        v16h ar = load_frag_A(s_mat, 0, wave * 32, LDL, lane);
        v16h br = load_frag_B(Ww1T, 0, wave * 32, CCH, lane);
        v8f acc2;
        for (int r = 0; r < 8; ++r) acc2[r] = 0.f;
        acc2 = wmma_f16(ar, br, acc2);
        for (int r = 0; r < 8; ++r)
            s_part[wave * 256 + (r + 8 * hi) * 16 + ncol] = acc2[r];
    }
    __syncthreads();

    {
        int g = t & 15;
        float sum = bw1[g];
        for (int w = 0; w < 8; ++w) sum += s_part[w * 256 + t];
        wpre[(size_t)point * 256 + t] = sum;
        atomicAdd(&s_sum[g], sum);
        atomicAdd(&s_ssq[g], sum * sum);
    }
    __syncthreads();
    if (t < 16) {
        atomicAdd(&wstats[t],      s_sum[t]);
        atomicAdd(&wstats[16 + t], s_ssq[t]);
    }
}

__global__ void finalize_w_kernel(
    const float* __restrict__ wstats, const float* __restrict__ gw,
    const float* __restrict__ Bw, float* __restrict__ bnw)
{
    int g = threadIdx.x;
    if (g >= GRP) return;
    float cnt = (float)NPTS * (float)SNB;
    float m = wstats[g] / cnt;
    float var = wstats[16 + g] / cnt - m * m;
    float a = gw[g] * rsqrtf(var + EPSV);
    bnw[g]      = a;
    bnw[16 + g] = Bw[g] - m * a;
}

// ---------------------------------------------------------------------------
// Pass 3: weight head (BN->relu->@Ww2->softmax*mask) + recompute peb (WMMA)
//          + value = v[idx]+peb, grouped weighted sum -> out[point]
// ---------------------------------------------------------------------------
__global__ __launch_bounds__(256) void output_kernel(
    const float* __restrict__ coord, const int* __restrict__ ref,
    const float* __restrict__ v_in, const float* __restrict__ wpre,
    const float* __restrict__ Wp1, const float* __restrict__ bp2,
    const _Float16* __restrict__ Wp2T,
    const float* __restrict__ Ww2, const float* __restrict__ bw2,
    const float* __restrict__ bnp, const float* __restrict__ bnw,
    float* __restrict__ out)
{
    const int LDL = 264;
    __shared__ __align__(16) _Float16 s_mat[16 * LDL];
    __shared__ float s_h[256];
    __shared__ float s_l[256];
    __shared__ float s_w[256];
    __shared__ float s_pos[16][3];
    __shared__ int   s_idx[16];
    __shared__ float s_mask[16];

    int t = threadIdx.x;
    int point = blockIdx.x;
    if (t < 16) {
        int r = ref[(size_t)point * SNB + t];
        s_mask[t] = (r >= 0) ? 1.f : 0.f;
        int idx = r < 0 ? 0 : r;
        s_idx[t] = idx;
        float cx = coord[(size_t)point * 3 + 0];
        float cy = coord[(size_t)point * 3 + 1];
        float cz = coord[(size_t)point * 3 + 2];
        s_pos[t][0] = coord[(size_t)idx * 3 + 0] - cx;
        s_pos[t][1] = coord[(size_t)idx * 3 + 1] - cy;
        s_pos[t][2] = coord[(size_t)idx * 3 + 2] - cz;
    }
    __syncthreads();

    {   // h = relu(BN(wpre))
        int g = t & 15;
        float x = wpre[(size_t)point * 256 + t] * bnw[g] + bnw[16 + g];
        s_h[t] = x > 0.f ? x : 0.f;
    }
    __syncthreads();
    {   // logits = h @ Ww2 + bw2
        int s = t >> 4, g = t & 15;
        float acc = bw2[g];
        for (int j = 0; j < GRP; ++j) acc += s_h[s * 16 + j] * Ww2[j * GRP + g];
        s_l[t] = acc;
    }
    __syncthreads();
    {   // softmax over neighbors, then mask
        int s = t >> 4, g = t & 15;
        float mx = -1e30f;
        for (int j = 0; j < SNB; ++j) mx = fmaxf(mx, s_l[j * 16 + g]);
        float den = 0.f;
        for (int j = 0; j < SNB; ++j) den += __expf(s_l[j * 16 + g] - mx);
        s_w[t] = s_mask[s] * __expf(s_l[t] - mx) / den;
    }
    __syncthreads();

    {   // pe1 recompute
        int c = t;
        float w0 = Wp1[c], w1 = Wp1[CCH + c], w2 = Wp1[2 * CCH + c];
        float a = bnp[1024 + c], b = bnp[1280 + c];
        for (int s = 0; s < SNB; ++s) {
            float z = s_pos[s][0] * w0 + s_pos[s][1] * w1 + s_pos[s][2] * w2;
            float val = z * a + b;
            s_mat[s * LDL + c] = (_Float16)(val > 0.f ? val : 0.f);
        }
    }
    __syncthreads();

    int wave = t >> 5, lane = t & 31;
    int ncol = lane & 15, hi = lane >> 4;
    v16h a[8];
    for (int kk = 0; kk < 8; ++kk)
        a[kk] = load_frag_A(s_mat, 0, kk * 32, LDL, lane);

    int n0a = wave * 32, n0b = n0a + 16;
    v8f acc0, acc1;
    { float bv = bp2[n0a + ncol]; for (int r = 0; r < 8; ++r) acc0[r] = bv; }
    { float bv = bp2[n0b + ncol]; for (int r = 0; r < 8; ++r) acc1[r] = bv; }
    for (int kk = 0; kk < 8; ++kk) {
        acc0 = wmma_f16(a[kk], load_frag_B(Wp2T, n0a, kk * 32, CCH, lane), acc0);
        acc1 = wmma_f16(a[kk], load_frag_B(Wp2T, n0b, kk * 32, CCH, lane), acc1);
    }

    {   // out[n] = sum_s w[s][n>>4] * (v[idx[s]][n] + peb[s][n])
        int n = n0a + ncol;
        float part = 0.f;
        for (int r = 0; r < 8; ++r) {
            int m = r + 8 * hi;
            float val = acc0[r] + v_in[(size_t)s_idx[m] * CCH + n];
            part += s_w[m * 16 + (n >> 4)] * val;
        }
        part += __shfl_xor(part, 16, 32);
        if (hi == 0) out[(size_t)point * CCH + n] = part;

        n = n0b + ncol;
        part = 0.f;
        for (int r = 0; r < 8; ++r) {
            int m = r + 8 * hi;
            float val = acc1[r] + v_in[(size_t)s_idx[m] * CCH + n];
            part += s_w[m * 16 + (n >> 4)] * val;
        }
        part += __shfl_xor(part, 16, 32);
        if (hi == 0) out[(size_t)point * CCH + n] = part;
    }
}

// ---------------------------------------------------------------------------
extern "C" void kernel_launch(void* const* d_in, const int* in_sizes, int n_in,
                              void* d_out, int out_size, void* d_ws, size_t ws_size,
                              hipStream_t stream)
{
    (void)in_sizes; (void)n_in; (void)out_size; (void)ws_size;

    const float* feat  = (const float*)d_in[0];
    const float* coord = (const float*)d_in[1];
    const int*   ref   = (const int*)  d_in[2];
    const float* Wq  = (const float*)d_in[3];
    const float* bq  = (const float*)d_in[4];
    const float* gq  = (const float*)d_in[5];
    const float* Bq  = (const float*)d_in[6];
    const float* Wk  = (const float*)d_in[7];
    const float* bk  = (const float*)d_in[8];
    const float* gk  = (const float*)d_in[9];
    const float* Bk  = (const float*)d_in[10];
    const float* Wv  = (const float*)d_in[11];
    const float* bv  = (const float*)d_in[12];
    const float* Wp1 = (const float*)d_in[13];
    /* bp1 (d_in[14]) cancels inside BN -> unused */
    const float* gp  = (const float*)d_in[15];
    const float* Bp  = (const float*)d_in[16];
    /* Wp2 */  const float* Wp2 = (const float*)d_in[17];
    const float* bp2 = (const float*)d_in[18];
    const float* Ww1 = (const float*)d_in[19];
    const float* bw1 = (const float*)d_in[20];
    const float* gw  = (const float*)d_in[21];
    const float* Bw  = (const float*)d_in[22];
    const float* Ww2 = (const float*)d_in[23];
    const float* bw2 = (const float*)d_in[24];
    float* out = (float*)d_out;

    // workspace layout
    const size_t NC = (size_t)NPTS * CCH;
    char* ws = (char*)d_ws;
    _Float16* feat16 = (_Float16*)ws;                        // 32 MB
    _Float16* WqT    = (_Float16*)(ws + NC * 2);
    _Float16* WkT    = WqT  + (size_t)CCH * CCH;
    _Float16* WvT    = WkT  + (size_t)CCH * CCH;
    _Float16* Wp2T   = WvT  + (size_t)CCH * CCH;
    _Float16* Ww1T   = Wp2T + (size_t)CCH * CCH;
    char* p = (char*)(Ww1T + (size_t)GRP * CCH);             // 34,086,912 bytes in
    float* q_lin = (float*)p;                                // 64 MB
    float* k_lin = q_lin + NC;                               // 64 MB
    float* v_buf = k_lin + NC;                               // 64 MB
    float* wpre  = v_buf + NC;                               // 64 MB (N*S*G)
    float* stats  = wpre + NC;       // [0..1023]  q/k sums
    float* pstats = stats + 1024;    // [1024..1039] pos 3x3 stats
    float* wstats = stats + 1040;    // [1040..1071] wpre sums
    float* bnp    = stats + 1072;    // 1536 floats: aQ,bQ,aK,bK,aP,cP
    float* bnw    = stats + 2608;    // 32 floats: aW,bW

    // zero the atomic accumulators each call (deterministic across replays)
    hipMemsetAsync(stats, 0, 1072 * sizeof(float), stream);

    prep_kernel<<<NC / 256, 256, 0, stream>>>(feat, Wq, Wk, Wv, Wp2, Ww1,
                                              feat16, WqT, WkT, WvT, Wp2T, Ww1T);
    gemm_qkv_kernel<<<NPTS / 128, 256, 0, stream>>>(feat16, WqT, WkT, WvT,
                                                    bq, bk, bv, q_lin, k_lin, v_buf);
    channel_stats_kernel<<<NPTS / 256, 256, 0, stream>>>(q_lin, k_lin, stats);
    pos_stats_kernel<<<(NPTS * SNB) / 256, 256, 0, stream>>>(coord, ref, pstats);
    finalize_bn_kernel<<<1, 256, 0, stream>>>(stats, pstats, gq, Bq, gk, Bk,
                                              Wp1, gp, Bp, bnp);
    apply_bn_kernel<<<NC / 256, 256, 0, stream>>>(q_lin, k_lin, bnp);
    wpre_kernel<<<NPTS, 256, 0, stream>>>(coord, ref, q_lin, k_lin, Wp1, bp2,
                                          Wp2T, Ww1T, bw1, bnp, wpre, wstats);
    finalize_w_kernel<<<1, 32, 0, stream>>>(wstats, gw, Bw, bnw);
    output_kernel<<<NPTS, 256, 0, stream>>>(coord, ref, v_buf, wpre, Wp1, bp2,
                                            Wp2T, Ww2, bw2, bnp, bnw, out);
}
